// DilatedAttentionBlock_attention_19473381720369
// MI455X (gfx1250) — matrix operand order repaired and checked by reading, not compiled, not run on hardware
//
#include <hip/hip_runtime.h>
#include <math.h>

// Problem constants (match reference)
#define Bn  8
#define Cn  256
#define Ln  2048
#define Hn  4
#define HDn 64
#define TOKn (Bn * Ln)   // 16384 tokens

typedef __bf16          v16bf  __attribute__((ext_vector_type(16)));
typedef float           v8f    __attribute__((ext_vector_type(8)));
typedef unsigned short  ushx8  __attribute__((ext_vector_type(8)));

union BF16Frag {
    v16bf bf;
    struct { ushx8 lo, hi; } v;
    unsigned short s[16];
};

// float -> bf16 bits, round to nearest even
__device__ inline unsigned short f2bf(float f) {
    unsigned int u = __builtin_bit_cast(unsigned int, f);
    if ((u & 0x7fffffffu) > 0x7f800000u)       // NaN
        return (unsigned short)((u >> 16) | 0x40);
    u += 0x7fffu + ((u >> 16) & 1u);
    return (unsigned short)(u >> 16);
}

__device__ inline v8f wmma_bf16(v16bf a, v16bf b, v8f c) {
    // v_wmma_f32_16x16x32_bf16  (neg_a, A, neg_b, B, c_mod, C, reuse_a, reuse_b)
    return __builtin_amdgcn_wmma_f32_16x16x32_bf16(false, a, false, b,
                                                   (short)0, c, false, false);
}

// CDNA5 async global->LDS DMA (ASYNCcnt), 16 bytes per enabled lane.
// Generic LDS pointer: low 32 bits == LDS byte offset (ISA aperture rule).
__device__ inline void async_copy16(const void* gptr, void* lds) {
    unsigned l = (unsigned)(size_t)lds;
    asm volatile("global_load_async_to_lds_b128 %0, %1, off"
                 :: "v"(l), "v"(gptr) : "memory");
}
__device__ inline void async_wait0() {
    asm volatile("s_wait_asynccnt 0x0" ::: "memory");
}

// A-matrix fragment (16x32 bf16), logical A[m,k] at base[(r0+m)*ld + k0+k].
// Layout (ISA 7.12.2): lane<16: row=lane, k={0..7,16..23}; lane>=16: row=lane-16, k={8..15,24..31}
__device__ inline v16bf load_frag_a16(const unsigned short* base, int r0, int k0, int ld) {
    const int lane = threadIdx.x & 31;
    const unsigned short* p = base + (size_t)(r0 + (lane & 15)) * ld + k0 + ((lane >> 4) << 3);
    BF16Frag fr;
    fr.v.lo = *(const ushx8*)(p);
    fr.v.hi = *(const ushx8*)(p + 16);
    return fr.bf;
}

// B-matrix fragment (32x16 bf16) from column-contiguous storage:
//   logical B[k,n] = base[(n0+n)*ld + k0+k]   (transposed weights W^T, key rows
//   K, transposed values V^T).  Per-lane: k={0..7,16..23} (lane<16) / {8..15,24..31} -> two 16B loads.
__device__ inline v16bf load_frag_bt16(const unsigned short* base, int n0, int k0, int ld) {
    const int lane = threadIdx.x & 31;
    const unsigned short* p = base + (size_t)(n0 + (lane & 15)) * ld + k0 + ((lane >> 4) << 3);
    BF16Frag fr;
    fr.v.lo = *(const ushx8*)(p);
    fr.v.hi = *(const ushx8*)(p + 16);
    return fr.bf;
}

// ---------------------------------------------------------------------------
// Kernel 1: fp32 -> bf16 weight conversion + transpose (7 matrices 256x256).
// dst holds W^T: dst[n*256 + k] = W[k*256 + n]  (writes coalesced).
// ---------------------------------------------------------------------------
__global__ void wconv_kernel(const float* w0, const float* w1, const float* w2,
                             const float* w3, const float* w4, const float* w5,
                             const float* w6, unsigned short* dst) {
    int gid = blockIdx.x * blockDim.x + threadIdx.x;   // grid covers 7*65536 exactly
    int wi = gid >> 16, e = gid & 65535;
    int n = e >> 8, k = e & 255;
    const float* srcs[7] = {w0, w1, w2, w3, w4, w5, w6};
    dst[gid] = f2bf(srcs[wi][k * 256 + n]);
}

// ---------------------------------------------------------------------------
// Kernel 2: x_in [B,C,L] fp32 -> Xbf [B*L, C] bf16 (tiled transpose)
// ---------------------------------------------------------------------------
__global__ void transpose_kernel(const float* __restrict__ x, unsigned short* __restrict__ Xbf) {
    __shared__ float tile[32][33];
    const int b = blockIdx.z;
    const int l0 = blockIdx.x * 32, c0 = blockIdx.y * 32;
    for (int r = threadIdx.y; r < 32; r += 8)
        tile[r][threadIdx.x] = x[(size_t)b * Cn * Ln + (size_t)(c0 + r) * Ln + l0 + threadIdx.x];
    __syncthreads();
    for (int r = threadIdx.y; r < 32; r += 8)
        Xbf[((size_t)b * Ln + l0 + r) * Cn + c0 + threadIdx.x] = f2bf(tile[threadIdx.x][r]);
}

// ---------------------------------------------------------------------------
// Kernel 3: bf16 WMMA GEMM  Out = f(A[M,256] @ W + bias), W given TRANSPOSED
//   (Wt[n*256+k]).  do_gelu: exact GELU.  layout: 0 row-major [M,256],
//   1 head layout [B,H,L,HD], 2 transposed head layout [B,H,HD,L] (for V).
//   Block 256 threads = 8 waves (4x2), tile 128x128, K-step 32.
//   Double-buffered async global->LDS staging overlapped with WMMAs.
// ---------------------------------------------------------------------------
__global__ __launch_bounds__(256) void gemm_kernel(
    const unsigned short* __restrict__ A, const unsigned short* __restrict__ Wt,
    const float* __restrict__ bias, unsigned short* __restrict__ Out,
    int do_gelu, int layout) {
    __shared__ __align__(16) unsigned short As[2][128 * 32];    // [m][k]
    __shared__ __align__(16) unsigned short WsT[2][128 * 32];   // [n][k]
    const int tid = threadIdx.x, lane = tid & 31, wid = tid >> 5;
    const int wm = wid >> 1, wn = wid & 1;              // 4x2 wave grid
    const int m0 = blockIdx.x * 128, n0 = blockIdx.y * 128;

    // per-thread staging coordinates
    const int sar = tid >> 1, sac = (tid & 1) << 4;     // A: 128 rows x 32 k
    const int swr = tid >> 2, swc = (tid & 3) << 3;     // Wt: 4 thr/row, 2 rows

    auto stage = [&](int buf, int kk) {
        const unsigned short* srcA = A + (size_t)(m0 + sar) * 256 + kk + sac;
        async_copy16(srcA,     &As[buf][sar * 32 + sac]);
        async_copy16(srcA + 8, &As[buf][sar * 32 + sac + 8]);
        async_copy16(Wt + (size_t)(n0 + swr) * 256 + kk + swc,
                     &WsT[buf][swr * 32 + swc]);
        async_copy16(Wt + (size_t)(n0 + swr + 64) * 256 + kk + swc,
                     &WsT[buf][(swr + 64) * 32 + swc]);
    };

    v8f acc[2][4] = {};
    stage(0, 0);
    async_wait0();
    __syncthreads();
    int buf = 0;
    for (int kk = 0; kk < 256; kk += 32) {
        if (kk + 32 < 256) stage(buf ^ 1, kk + 32);     // prefetch next k-tile
        v16bf af0 = load_frag_a16(As[buf], wm * 32, 0, 32);
        v16bf af1 = load_frag_a16(As[buf], wm * 32 + 16, 0, 32);
#pragma unroll
        for (int ni = 0; ni < 4; ++ni) {
            v16bf bfr = load_frag_bt16(WsT[buf], wn * 64 + ni * 16, 0, 32);
            acc[0][ni] = wmma_bf16(af0, bfr, acc[0][ni]);
            acc[1][ni] = wmma_bf16(af1, bfr, acc[1][ni]);
        }
        async_wait0();
        __syncthreads();
        buf ^= 1;
    }
    const int rhalf = (lane >> 4) << 3;
#pragma unroll
    for (int mi = 0; mi < 2; ++mi)
#pragma unroll
        for (int ni = 0; ni < 4; ++ni)
#pragma unroll
            for (int i = 0; i < 8; ++i) {
                int m = m0 + wm * 32 + mi * 16 + rhalf + i;
                int n = n0 + wn * 64 + ni * 16 + (lane & 15);
                float v = acc[mi][ni][i] + bias[n];
                if (do_gelu) v = 0.5f * v * (1.0f + erff(v * 0.70710678118654752f));
                unsigned short bv = f2bf(v);
                int b = m >> 11, l = m & 2047, h = n >> 6, hd = n & 63;
                if (layout == 0)
                    Out[(size_t)m * 256 + n] = bv;
                else if (layout == 1)   // [B,H,L,HD]
                    Out[(((size_t)(b * Hn + h)) * Ln + l) * HDn + hd] = bv;
                else                    // [B,H,HD,L]  (V transposed)
                    Out[(((size_t)(b * Hn + h)) * HDn + hd) * Ln + l] = bv;
            }
}

// ---------------------------------------------------------------------------
// Kernel 4: attention per (b,h). Block 128 = 4 waves, each wave owns 16 query
// rows. Pass 1: row sum-of-squares of scores (WMMA QK^T). Scaled scores lie in
// [-1,1] after L2 rescale -> softmax without max subtraction. Pass 2:
// recompute scores, exp, accumulate row-sum and P@V (WMMA; V is pre-transposed
// so its fragments are contiguous; P staged via wave-private LDS).
// ---------------------------------------------------------------------------
__global__ __launch_bounds__(128) void attn_kernel(
    const unsigned short* __restrict__ Q, const unsigned short* __restrict__ K,
    const unsigned short* __restrict__ Vt, unsigned short* __restrict__ O) {
    __shared__ __align__(16) unsigned short P[4][16 * 32];
    const int tid = threadIdx.x, lane = tid & 31, wid = tid >> 5;
    const int bh = blockIdx.y;
    const int q0 = blockIdx.x * 64 + wid * 16;
    const size_t base = (size_t)bh * Ln * HDn;
    const unsigned short* Qh  = Q  + base;   // [L, HD]
    const unsigned short* Kh  = K  + base;   // [L, HD]
    const unsigned short* Vth = Vt + base;   // [HD, L]

    const v16bf aq0 = load_frag_a16(Qh, q0, 0, HDn);
    const v16bf aq1 = load_frag_a16(Qh, q0, 32, HDn);

    // ---- pass 1: sum of squares of raw scores (scale folded in later) ----
    v8f ssq = {};
    for (int kb = 0; kb < Ln; kb += 16) {
        v8f s = {};
        s = wmma_bf16(aq0, load_frag_bt16(Kh, kb, 0, HDn), s);
        s = wmma_bf16(aq1, load_frag_bt16(Kh, kb, 32, HDn), s);
        ssq += s * s;
    }
#pragma unroll
    for (int i = 0; i < 8; ++i) {   // reduce across the 16-lane half (cols)
        float v = ssq[i];
        v += __shfl_xor(v, 1, 32);
        v += __shfl_xor(v, 2, 32);
        v += __shfl_xor(v, 4, 32);
        v += __shfl_xor(v, 8, 32);
        ssq[i] = v;
    }
    // combined per-row factor on raw acc: (1/8) / (sqrt(ssq/64)*LIP + EPS)
    v8f fsc;
#pragma unroll
    for (int i = 0; i < 8; ++i)
        fsc[i] = 0.125f / (sqrtf(ssq[i] * (1.0f / 64.0f)) + 1e-12f);

    // ---- pass 2: streaming softmax + P@V ----
    v8f rs = {};
    v8f o[4] = {};
    unsigned short* Pw = P[wid];
    const int col = lane & 15;
    const int rhalf = (lane >> 4) << 3;
    for (int kb = 0; kb < Ln; kb += 32) {
        v8f s0 = {}, s1 = {};
        s0 = wmma_bf16(aq0, load_frag_bt16(Kh, kb, 0, HDn), s0);
        s0 = wmma_bf16(aq1, load_frag_bt16(Kh, kb, 32, HDn), s0);
        s1 = wmma_bf16(aq0, load_frag_bt16(Kh, kb + 16, 0, HDn), s1);
        s1 = wmma_bf16(aq1, load_frag_bt16(Kh, kb + 16, 32, HDn), s1);
#pragma unroll
        for (int i = 0; i < 8; ++i) {
            float p0 = __expf(s0[i] * fsc[i]);   // args in [-1,1]: safe
            float p1 = __expf(s1[i] * fsc[i]);
            rs[i] += p0 + p1;
            Pw[(rhalf + i) * 32 + col]      = f2bf(p0);   // C-layout -> LDS tile
            Pw[(rhalf + i) * 32 + 16 + col] = f2bf(p1);
        }
        v16bf ap = load_frag_a16(Pw, 0, 0, 32);  // wave-private LDS, DS in-order
#pragma unroll
        for (int ni = 0; ni < 4; ++ni)
            o[ni] = wmma_bf16(ap, load_frag_bt16(Vth, ni * 16, kb, Ln), o[ni]);
    }
#pragma unroll
    for (int i = 0; i < 8; ++i) {
        float v = rs[i];
        v += __shfl_xor(v, 1, 32);
        v += __shfl_xor(v, 2, 32);
        v += __shfl_xor(v, 4, 32);
        v += __shfl_xor(v, 8, 32);
        rs[i] = 1.0f / v;
    }
    const int b = bh >> 2, h = bh & 3;
#pragma unroll
    for (int ni = 0; ni < 4; ++ni)
#pragma unroll
        for (int i = 0; i < 8; ++i) {
            int q = q0 + rhalf + i;
            int c = h * HDn + ni * 16 + col;
            O[((size_t)(b * Ln + q)) * Cn + c] = f2bf(o[ni][i] * rs[i]);
        }
}

// ---------------------------------------------------------------------------
// Kernel 5: out = elu(LN(Attn @ o_w + o_b)) transposed to [B,C,L], + x_in.
// o_w passed TRANSPOSED. Block 256 = 8 waves (2x4), 32 rows x 256 cols.
// Async single-buffered staging (LDS budget holds the fp32 Y tile).
// ---------------------------------------------------------------------------
__global__ __launch_bounds__(256) void out_kernel(
    const unsigned short* __restrict__ A, const unsigned short* __restrict__ Wt,
    const float* __restrict__ ob, const float* __restrict__ lng,
    const float* __restrict__ lnb, const float* __restrict__ xin,
    float* __restrict__ out) {
    __shared__ __align__(16) unsigned short As[32 * 32];    // [m][k]
    __shared__ __align__(16) unsigned short WsT[256 * 32];  // [n][k]
    __shared__ float Y[32 * 256];
    __shared__ float MU[32], RS[32];
    const int tid = threadIdx.x, lane = tid & 31, wid = tid >> 5;
    const int wm = wid >> 2, wn = wid & 3;
    const int m0 = blockIdx.x * 32;

    v8f acc[4] = {};
    for (int kk = 0; kk < 256; kk += 32) {
        if (tid < 128) {   // stage A tile 32x32 (async DMA)
            int r = tid >> 2, c = (tid & 3) << 3;
            async_copy16(A + (size_t)(m0 + r) * 256 + kk + c, &As[r * 32 + c]);
        }
        {   // stage Wt tile 256(n) x 32(k): 4 threads per n-row, 4 rows each
            int rbase = tid >> 2, c = (tid & 3) << 3;
#pragma unroll
            for (int rr = 0; rr < 4; ++rr) {
                int n = rr * 64 + rbase;
                async_copy16(Wt + (size_t)n * 256 + kk + c, &WsT[n * 32 + c]);
            }
        }
        async_wait0();
        __syncthreads();
        v16bf af = load_frag_a16(As, wm * 16, 0, 32);
#pragma unroll
        for (int ni = 0; ni < 4; ++ni)
            acc[ni] = wmma_bf16(af, load_frag_bt16(WsT, wn * 64 + ni * 16, 0, 32), acc[ni]);
        __syncthreads();
    }
    const int rhalf = (lane >> 4) << 3;
#pragma unroll
    for (int ni = 0; ni < 4; ++ni)
#pragma unroll
        for (int i = 0; i < 8; ++i) {
            int m = wm * 16 + rhalf + i;
            int n = wn * 64 + ni * 16 + (lane & 15);
            Y[m * 256 + n] = acc[ni][i] + ob[n];
        }
    __syncthreads();
    if (tid < 32) {   // per-row LayerNorm stats
        float s = 0.f;
        for (int c = 0; c < 256; ++c) s += Y[tid * 256 + c];
        float mu = s * (1.0f / 256.0f), v = 0.f;
        for (int c = 0; c < 256; ++c) { float d = Y[tid * 256 + c] - mu; v += d * d; }
        MU[tid] = mu;
        RS[tid] = rsqrtf(v * (1.0f / 256.0f) + 1e-5f);
    }
    __syncthreads();
    for (int idx = tid; idx < 32 * 256; idx += 256) {
        int m = idx & 31, c = idx >> 5;        // consecutive tid -> consecutive l
        float y = (Y[m * 256 + c] - MU[m]) * RS[m] * lng[c] + lnb[c];
        float e = (y > 0.f) ? y : expm1f(y);   // ELU(alpha=1)
        int tok = m0 + m, b = tok >> 11, l = tok & 2047;
        size_t gi = ((size_t)b * Cn + c) * Ln + l;
        out[gi] = e + xin[gi];
    }
}

// ---------------------------------------------------------------------------
extern "C" void kernel_launch(void* const* d_in, const int* in_sizes, int n_in,
                              void* d_out, int out_size, void* d_ws, size_t ws_size,
                              hipStream_t stream) {
    const float* x_in = (const float*)d_in[0];
    const float* qw1 = (const float*)d_in[1];  const float* qb1 = (const float*)d_in[2];
    const float* qw2 = (const float*)d_in[3];  const float* qb2 = (const float*)d_in[4];
    const float* kw1 = (const float*)d_in[5];  const float* kb1 = (const float*)d_in[6];
    const float* kw2 = (const float*)d_in[7];  const float* kb2 = (const float*)d_in[8];
    const float* vw1 = (const float*)d_in[9];  const float* vb1 = (const float*)d_in[10];
    const float* vw2 = (const float*)d_in[11]; const float* vb2 = (const float*)d_in[12];
    const float* ow  = (const float*)d_in[13]; const float* obi = (const float*)d_in[14];
    const float* lng = (const float*)d_in[15]; const float* lnb = (const float*)d_in[16];

    // Workspace layout (~49 MB): 6 bf16 token buffers of 8 MB + bf16 W^T.
    char* ws = (char*)d_ws;
    unsigned short* Xbf = (unsigned short*)(ws + (size_t)0);
    unsigned short* Hid = (unsigned short*)(ws + ((size_t)8  << 20));
    unsigned short* Qb  = (unsigned short*)(ws + ((size_t)16 << 20));
    unsigned short* Kb  = (unsigned short*)(ws + ((size_t)24 << 20));
    unsigned short* Vb  = (unsigned short*)(ws + ((size_t)32 << 20));  // transposed [B,H,HD,L]
    unsigned short* Att = (unsigned short*)(ws + ((size_t)40 << 20));
    unsigned short* Wb  = (unsigned short*)(ws + ((size_t)48 << 20));  // 7 x 65536 (W^T)

    wconv_kernel<<<7 * 65536 / 256, 256, 0, stream>>>(qw1, qw2, kw1, kw2, vw1, vw2, ow, Wb);
    transpose_kernel<<<dim3(Ln / 32, Cn / 32, Bn), dim3(32, 8), 0, stream>>>(x_in, Xbf);

    dim3 gg(TOKn / 128, 2);
    gemm_kernel<<<gg, 256, 0, stream>>>(Xbf, Wb + 0 * 65536, qb1, Hid, 1, 0);
    gemm_kernel<<<gg, 256, 0, stream>>>(Hid, Wb + 1 * 65536, qb2, Qb,  0, 1);
    gemm_kernel<<<gg, 256, 0, stream>>>(Xbf, Wb + 2 * 65536, kb1, Hid, 1, 0);
    gemm_kernel<<<gg, 256, 0, stream>>>(Hid, Wb + 3 * 65536, kb2, Kb,  0, 1);
    gemm_kernel<<<gg, 256, 0, stream>>>(Xbf, Wb + 4 * 65536, vb1, Hid, 1, 0);
    gemm_kernel<<<gg, 256, 0, stream>>>(Hid, Wb + 5 * 65536, vb2, Vb,  0, 2);

    attn_kernel<<<dim3(Ln / 64, Bn * Hn), 128, 0, stream>>>(Qb, Kb, Vb, Att);

    out_kernel<<<TOKn / 32, 256, 0, stream>>>(Att, Wb + 6 * 65536, obi, lng, lnb,
                                              x_in, (float*)d_out);
}
